// ANFSAT_58772332478676
// MI455X (gfx1250) — matrix operand order repaired and checked by reading, not compiled, not run on hardware
//
#include <hip/hip_runtime.h>
#include <hip/hip_bf16.h>
#include <math.h>

constexpr int FM = 128, QM = 128;
constexpr int NV = 8192, NA = 16384, NC = 32768, LITS = 3;
constexpr int NE = NC * LITS;      // 98304
constexpr int NR = 1 + NV + NA;    // 24577
constexpr int ROUNDS = 8;

typedef __attribute__((ext_vector_type(16))) __bf16         v16bf;
typedef __attribute__((ext_vector_type(8)))  float          v8f;
typedef __attribute__((ext_vector_type(8)))  unsigned short v8u;
typedef __attribute__((ext_vector_type(16))) unsigned short v16u;

union BF16x16 { v16u u; v16bf b; };

__device__ __forceinline__ float bf2f(unsigned short u) {
    unsigned int x = ((unsigned int)u) << 16; float f;
    __builtin_memcpy(&f, &x, 4); return f;
}
__device__ __forceinline__ unsigned short f2bf(float f) {
    unsigned int u; __builtin_memcpy(&u, &f, 4);
    unsigned int r = u + 0x7fffu + ((u >> 16) & 1u);
    return (unsigned short)(r >> 16);
}

// ---------------------------------------------------------------------------
// WMMA GEMM with LDS double-buffered staging.
//   Out[M,N] = act(A[M,K](bf16) * Wt[N,K](bf16)^T + bias)
// 256 thr = 8 waves. Block tile 64(M) x 128(N), K stage depth 64.
// Wave -> 16x64 strip -> 4 accumulators, 8 wmma per stage.
// Stage k+1 global loads are issued before stage-k WMMAs (latency hiding);
// all 4 B fragments live in distinct registers so ds_loads for a kk-step are
// issued as one burst with staged s_wait_dscnt, overlapping LDS latency with
// the WMMA pipe. Requires: M%64==0, N%128==0, K%64==0.
// ---------------------------------------------------------------------------
template<bool RELU, bool OUTBF>
__global__ __launch_bounds__(256)
void k_gemm(const unsigned short* __restrict__ A,
            const unsigned short* __restrict__ Wt,
            const float* __restrict__ bias,
            void* __restrict__ Out, int M, int K, int N)
{
    __shared__ unsigned short smA[2][64 * 64];    // [row][k] 8KB per buf
    __shared__ unsigned short smB[2][128 * 64];   // [col][k] 16KB per buf

    const int tid   = threadIdx.x;
    const int lane  = tid & 31;
    const int wave  = tid >> 5;
    const int waveM = wave & 3, waveN = wave >> 2;
    const int half  = lane >> 4;
    const int mloc  = waveM * 16 + (lane & 15);   // row inside A tile
    const int ncol  = lane & 15;
    const int kbA   = half ? 8 : 0;               // A 16x32 bf16 lane layout
    const int kbB   = half ? 8 : 0;               // B 32x16 bf16 lane layout

    // cooperative global->reg tile mapping
    const int ar = tid >> 2, ac = (tid & 3) << 4;   // A: 64x64, 16 elem/thr
    const int br = tid >> 1, bc = (tid & 1) << 5;   // B: 128x64, 32 elem/thr
    const unsigned short* gA = A  + (size_t)(blockIdx.y * 64  + ar) * K + ac;
    const unsigned short* gB = Wt + (size_t)(blockIdx.x * 128 + br) * K + bc;

    v8u ra0, ra1, rb0, rb1, rb2, rb3;
    auto ld_regs = [&](int k0) {
        ra0 = *(const v8u*)(gA + k0);
        ra1 = *(const v8u*)(gA + k0 + 8);
        rb0 = *(const v8u*)(gB + k0);
        rb1 = *(const v8u*)(gB + k0 + 8);
        rb2 = *(const v8u*)(gB + k0 + 16);
        rb3 = *(const v8u*)(gB + k0 + 24);
    };
    auto st_lds = [&](int buf) {
        *(v8u*)(&smA[buf][ar * 64 + ac])      = ra0;
        *(v8u*)(&smA[buf][ar * 64 + ac + 8])  = ra1;
        *(v8u*)(&smB[buf][br * 64 + bc])      = rb0;
        *(v8u*)(&smB[buf][br * 64 + bc + 8])  = rb1;
        *(v8u*)(&smB[buf][br * 64 + bc + 16]) = rb2;
        *(v8u*)(&smB[buf][br * 64 + bc + 24]) = rb3;
    };

    v8f c[4] = {v8f{}, v8f{}, v8f{}, v8f{}};
    ld_regs(0);
    st_lds(0);
    __syncthreads();

    int cur = 0;
    for (int k0 = 0; k0 < K; k0 += 64) {
        const bool hasNext = (k0 + 64) < K;
        if (hasNext) ld_regs(k0 + 64);            // in flight during WMMAs
#pragma unroll
        for (int kk = 0; kk < 64; kk += 32) {
            // ---- load A fragment + all 4 B fragments (distinct registers),
            //      then fire the 4 WMMAs as a burst.
            BF16x16 a, bm[4];
            {
                v8u alo = *(const v8u*)(&smA[cur][mloc * 64 + kk + kbA]);
                v8u ahi = *(const v8u*)(&smA[cur][mloc * 64 + kk + kbA + 16]);
#pragma unroll
                for (int i = 0; i < 8; ++i) { a.u[i] = alo[i]; a.u[i + 8] = ahi[i]; }
            }
#pragma unroll
            for (int t = 0; t < 4; ++t) {
                const unsigned short* bp =
                    &smB[cur][(waveN * 64 + t * 16 + ncol) * 64 + kk + kbB];
                v8u blo = *(const v8u*)(bp);
                v8u bhi = *(const v8u*)(bp + 16);
#pragma unroll
                for (int i = 0; i < 8; ++i) { bm[t].u[i] = blo[i]; bm[t].u[i + 8] = bhi[i]; }
            }
#pragma unroll
            for (int t = 0; t < 4; ++t)
                c[t] = __builtin_amdgcn_wmma_f32_16x16x32_bf16(
                           false, a.b, false, bm[t].b, (short)0, c[t], false, false);
        }
        if (hasNext) {
            __syncthreads();
            st_lds(cur ^ 1);     // s_wait_loadcnt lands here, after the WMMAs
            __syncthreads();
            cur ^= 1;
        }
    }

    const int tm  = blockIdx.y * 64  + waveM * 16;
    const int tn0 = blockIdx.x * 128 + waveN * 64;
#pragma unroll
    for (int t = 0; t < 4; ++t) {
        int col = tn0 + t * 16 + ncol;
        float bb = bias ? bias[col] : 0.f;
#pragma unroll
        for (int v = 0; v < 8; ++v) {
            int row = tm + v + (half << 3);
            float x = c[t][v] + bb;
            if (RELU) x = fmaxf(x, 0.f);
            if (OUTBF) ((unsigned short*)Out)[(size_t)row * N + col] = f2bf(x);
            else       ((float*)Out)[(size_t)row * N + col] = x;
        }
    }
}

// ---------------------------------------------------------------------------
// Weight convert+transpose: W f32 [din,dout] -> Wt bf16 [dout, dinP] (0-pad K)
// ---------------------------------------------------------------------------
__global__ void k_wt(const float* __restrict__ W, unsigned short* __restrict__ Wt,
                     int din, int dout, int dinP)
{
    long long i = blockIdx.x * 256LL + threadIdx.x;
    if (i >= (long long)dout * dinP) return;
    int n = (int)(i / dinP), k = (int)(i % dinP);
    Wt[i] = f2bf(k < din ? W[(size_t)k * dout + n] : 0.f);
}

// ------------------------- elementwise / pack kernels ----------------------
__device__ __forceinline__ unsigned hsh(unsigned x) {
    x ^= x >> 16; x *= 0x7feb352du; x ^= x >> 15; x *= 0x846ca68bu; x ^= x >> 16;
    return x;
}
__device__ __forceinline__ float gnoise(int step, int r, int c) {
    unsigned s1 = hsh(((unsigned)(r * 4 + c)) ^ (0x9E3779B9u * (unsigned)(step + 1)));
    unsigned s2 = hsh(s1 ^ 0x85ebca6bu);
    float u1 = (float)s1 * 2.3283064e-10f + 1e-9f;
    float u2 = (float)s2 * 2.3283064e-10f;
    return sqrtf(-2.f * logf(u1)) * cosf(6.28318530718f * u2);
}

__global__ void k_init_state(float* __restrict__ x, long long total) {
    long long i = blockIdx.x * 256LL + threadIdx.x;
    if (i >= total) return;
    int c = (int)(i & 127);
    x[i] = (((c == 0) ? 1.f : 0.f) - 0.0078125f) * 2.82842712f; // sqrt(128)*0.25
}
__global__ void k_count_deg(const int* __restrict__ rows, const int* __restrict__ cols,
                            float* __restrict__ vd, float* __restrict__ cd) {
    int e = blockIdx.x * 256 + threadIdx.x;
    if (e >= NE) return;
    atomicAdd(&vd[rows[e]], 1.f);
    atomicAdd(&cd[cols[e]], 1.f);
}
__global__ void k_rsqrt_deg(float* __restrict__ d, int n) {
    int i = blockIdx.x * 256 + threadIdx.x;
    if (i < n) d[i] = rsqrtf(fmaxf(d[i], 1.f));
}
__global__ void k_pack_qin(const float* __restrict__ vars, unsigned short* __restrict__ q,
                           int step) {
    long long i = blockIdx.x * 256LL + threadIdx.x;
    if (i >= (long long)NV * 192) return;
    int r = (int)(i / 192), c = (int)(i % 192);
    float v = (c < 128) ? vars[(size_t)r * 128 + c]
            : (c < 132) ? gnoise(step, r, c - 128) : 0.f;
    q[i] = f2bf(v);
}
__global__ void k_sigmoid(float* __restrict__ x, long long total) {
    long long i = blockIdx.x * 256LL + threadIdx.x;
    if (i < total) x[i] = 1.f / (1.f + expf(-x[i]));
}
__global__ void k_pands(const float* __restrict__ p, const int* __restrict__ i1,
                        const int* __restrict__ i2, float* __restrict__ pa) {
    long long i = blockIdx.x * 256LL + threadIdx.x;
    if (i >= (long long)NA * 128) return;
    int j = (int)(i >> 7), c = (int)(i & 127);
    pa[i] = p[(size_t)(i1[j] - 1) * 128 + c] * p[(size_t)(i2[j] - 1) * 128 + c];
}
__global__ void k_scatter_S(const int* __restrict__ rows, const int* __restrict__ cols,
                            const float* __restrict__ p, const float* __restrict__ pa,
                            float* __restrict__ S) {
    long long i = blockIdx.x * 256LL + threadIdx.x;
    if (i >= (long long)NE * 128) return;
    int e = (int)(i >> 7), c = (int)(i & 127);
    int r = rows[e];
    float v = (r == 0) ? 1.f
            : (r <= NV) ? p[(size_t)(r - 1) * 128 + c]
                        : pa[(size_t)(r - NV - 1) * 128 + c];
    atomicAdd(&S[(size_t)cols[e] * 128 + c], v);
}
__global__ void k_scatter_cmsg(const int* __restrict__ rows, const int* __restrict__ cols,
                               const float* __restrict__ zv, const float* __restrict__ vdat,
                               const float* __restrict__ adat, float* __restrict__ out) {
    long long i = blockIdx.x * 256LL + threadIdx.x;
    if (i >= (long long)NE * 128) return;
    int e = (int)(i >> 7), c = (int)(i & 127);
    int r = rows[e];
    float v = (r == 0) ? zv[c]
            : (r <= NV) ? vdat[(size_t)(r - 1) * 128 + c]
                        : adat[(size_t)(r - NV - 1) * 128 + c];
    atomicAdd(&out[(size_t)cols[e] * 128 + c], v);
}
__global__ void k_build_cu(const float* __restrict__ cs, const float* __restrict__ cm,
                           const float* __restrict__ cdw, const float* __restrict__ S,
                           unsigned short* __restrict__ cu) {
    long long i = blockIdx.x * 256LL + threadIdx.x;
    if (i >= (long long)NC * 128) return;
    int r = (int)(i >> 7), c = (int)(i & 127);
    unsigned short* row = cu + (size_t)r * 512;
    row[c]       = f2bf(cs[i]);
    row[128 + c] = f2bf(cm[i] * cdw[r]);
    float s = S[i] * 3.14159265359f;
    row[256 + c] = f2bf(cosf(s));
    row[384 + c] = f2bf(sinf(s));
}
__global__ void k_scatter_val(const int* __restrict__ rows, const int* __restrict__ cols,
                              const float* __restrict__ cdata, float* __restrict__ val) {
    long long i = blockIdx.x * 256LL + threadIdx.x;
    if (i >= (long long)NE * 128) return;
    int e = (int)(i >> 7), c = (int)(i & 127);
    atomicAdd(&val[(size_t)rows[e] * 128 + c], cdata[(size_t)cols[e] * 256 + c]);
}
__global__ void k_scale_val(float* __restrict__ val, const float* __restrict__ dw) {
    long long i = blockIdx.x * 256LL + threadIdx.x;
    if (i >= (long long)NR * 128) return;
    val[i] *= dw[i >> 7];
}
__global__ void k_pack_v2a(const float* __restrict__ vars, const int* __restrict__ i1,
                           const int* __restrict__ i2, unsigned short* __restrict__ out) {
    long long i = blockIdx.x * 256LL + threadIdx.x;
    if (i >= (long long)NA * 256) return;
    int j = (int)(i >> 8), c = (int)(i & 255);
    float v = (c < 128) ? vars[(size_t)(i1[j] - 1) * 128 + c]
                        : vars[(size_t)(i2[j] - 1) * 128 + (c - 128)];
    out[i] = f2bf(v);
}
__global__ void k_pack_am(const float* __restrict__ vars, const float* __restrict__ val,
                          const int* __restrict__ i1, const int* __restrict__ i2,
                          unsigned short* __restrict__ out) {
    long long i = blockIdx.x * 256LL + threadIdx.x;
    if (i >= (long long)NA * 384) return;
    int j = (int)(i / 384), c = (int)(i % 384);
    float v;
    if (c < 128)      v = vars[(size_t)(i1[j] - 1) * 128 + c];
    else if (c < 256) v = vars[(size_t)(i2[j] - 1) * 128 + (c - 128)];
    else              v = val[(size_t)(NV + 1 + j) * 128 + (c - 256)];
    out[i] = f2bf(v);
}
__global__ void k_scatter_ands(const float* __restrict__ av, const int* __restrict__ i1,
                               const int* __restrict__ i2, float* __restrict__ val) {
    long long i = blockIdx.x * 256LL + threadIdx.x;
    if (i >= (long long)NA * 128) return;
    int j = (int)(i >> 7), c = (int)(i & 127);
    float v = av[i];
    atomicAdd(&val[(size_t)i1[j] * 128 + c], v);   // var row = 1 + (idx-1)
    atomicAdd(&val[(size_t)i2[j] * 128 + c], v);
}
__global__ void k_pack_ug(const float* __restrict__ vars, const float* __restrict__ val,
                          unsigned short* __restrict__ out) {
    long long i = blockIdx.x * 256LL + threadIdx.x;
    if (i >= (long long)NV * 256) return;
    int r = (int)(i >> 8), c = (int)(i & 255);
    float v = (c < 128) ? vars[(size_t)r * 128 + c]
                        : val[(size_t)(1 + r) * 128 + (c - 128)];
    out[i] = f2bf(v);
}
__global__ void k_cvt_bf(const float* __restrict__ in, unsigned short* __restrict__ out,
                         long long total) {
    long long i = blockIdx.x * 256LL + threadIdx.x;
    if (i < total) out[i] = f2bf(in[i]);
}

// ------------------------------- PairNorm ----------------------------------
__global__ void k_pn_reduce(const float* __restrict__ x, int ld, int coff,
                            float* __restrict__ colsum, float* __restrict__ sumsq) {
    __shared__ float sm[128];
    int r = blockIdx.x, c = threadIdx.x;
    float v = x[(size_t)r * ld + coff + c];
    atomicAdd(&colsum[c], v);
    sm[c] = v * v; __syncthreads();
    for (int s = 64; s > 0; s >>= 1) { if (c < s) sm[c] += sm[c + s]; __syncthreads(); }
    if (c == 0) atomicAdd(sumsq, sm[0]);
}
__global__ void k_pn_stats(const float* __restrict__ colsum, const float* __restrict__ sumsq,
                           float* __restrict__ mean, float* __restrict__ rs, int n) {
    __shared__ float sm[128];
    int c = threadIdx.x;
    float m = colsum[c] / (float)n;
    mean[c] = m; sm[c] = m * m; __syncthreads();
    for (int s = 64; s > 0; s >>= 1) { if (c < s) sm[c] += sm[c + s]; __syncthreads(); }
    if (c == 0) {
        float var = (sumsq[0] - (float)n * sm[0]) / ((float)n * 128.f);
        rs[0] = rsqrtf(var + 1e-6f);
    }
}
__global__ void k_pn_apply(const float* __restrict__ x, int ld, int coff,
                           const float* __restrict__ mean, const float* __restrict__ rs,
                           float* __restrict__ state, long long total) {
    long long i = blockIdx.x * 256LL + threadIdx.x;
    if (i >= total) return;
    int r = (int)(i >> 7), c = (int)(i & 127);
    float pn = (x[(size_t)r * ld + coff + c] - mean[c]) * rs[0];
    state[i] = pn * 0.25f + 0.1f * state[i];
}

// --------------------------- output layer + loss ---------------------------
__global__ void k_out_final(const unsigned short* __restrict__ h2,
                            const float* __restrict__ w, const float* __restrict__ b,
                            const int* __restrict__ labels,
                            float* __restrict__ logits, float* __restrict__ loss_sum) {
    __shared__ float sm[256];
    int r = blockIdx.x * 256 + threadIdx.x;
    const unsigned short* hr = h2 + (size_t)r * 128;
    float acc = b[0];
#pragma unroll 8
    for (int k = 0; k < 128; ++k) acc += bf2f(hr[k]) * w[k];
    logits[r] = acc;
    float y = 0.05f + 0.9f * (float)labels[r];
    float l = fmaxf(acc, 0.f) - acc * y + log1pf(expf(-fabsf(acc)));
    sm[threadIdx.x] = l; __syncthreads();
    for (int s = 128; s > 0; s >>= 1) { if (threadIdx.x < s) sm[threadIdx.x] += sm[threadIdx.x + s]; __syncthreads(); }
    if (threadIdx.x == 0) atomicAdd(loss_sum, sm[0] / (float)NV);
}
__global__ void k_write_tail(const float* __restrict__ loss_sum, float* __restrict__ out) {
    if (threadIdx.x == 0) { out[NV] = loss_sum[0] / (float)ROUNDS; out[NV + 1] = (float)(ROUNDS - 1); }
}

// ===========================================================================
extern "C" void kernel_launch(void* const* d_in, const int* in_sizes, int n_in,
                              void* d_out, int out_size, void* d_ws, size_t ws_size,
                              hipStream_t stream)
{
    (void)in_sizes; (void)n_in; (void)out_size; (void)ws_size;
    // ---- inputs (params leaves in dict insertion order, then index arrays)
    const float* ZV   = (const float*)d_in[42];
    const int*   I1   = (const int*)d_in[43];
    const int*   I2   = (const int*)d_in[44];
    const int*   ROWS = (const int*)d_in[45];
    const int*   COLS = (const int*)d_in[46];
    const int*   LBL  = (const int*)d_in[47];
    float* out = (float*)d_out;

    // ---- workspace carve
    char* base = (char*)d_ws; size_t off = 0;
    auto alloc = [&](size_t bytes) -> void* {
        off = (off + 255) & ~(size_t)255; void* p = base + off; off += bytes; return p;
    };
    struct WInfo { int idx, din, dout, dinP; };
    static const WInfo wtab[20] = {
        {0,132,384,192},{2,384,384,384},{4,384,128,384},     // variables_query
        {6,256,256,256},{8,256,256,256},{10,256,128,256},    // var2ands
        {12,128,256,128},{14,256,256,256},{16,256,128,256},  // var2clause
        {18,512,256,512},{20,256,256,256},{22,256,256,256},  // clause_mlp
        {24,384,256,384},{26,256,256,256},{28,256,128,256},  // ands_mlp
        {30,256,256,256},{32,256,256,256},{34,256,128,256},  // update_gate
        {36,128,128,128},{38,128,128,128},                   // variables_output (L0,L1)
    };
    unsigned short* WT[20];
    for (int i = 0; i < 20; ++i)
        WT[i] = (unsigned short*)alloc((size_t)wtab[i].dout * wtab[i].dinP * 2);

    float* variables    = (float*)alloc((size_t)NV * 128 * 4);
    float* clause_state = (float*)alloc((size_t)NC * 128 * 4);
    float* qbuf         = (float*)alloc((size_t)NV * 128 * 4);   // query -> p
    float* p_ands       = (float*)alloc((size_t)NA * 128 * 4);
    float* Sbuf         = (float*)alloc((size_t)NC * 128 * 4);
    float* cmsg         = (float*)alloc((size_t)NC * 128 * 4);
    float* var_data     = (float*)alloc((size_t)NV * 128 * 4);
    float* ands_data    = (float*)alloc((size_t)NA * 128 * 4);
    float* clause_data  = (float*)alloc((size_t)NC * 256 * 4);
    float* val          = (float*)alloc((size_t)NR * 128 * 4);
    float* ands_value   = (float*)alloc((size_t)NA * 128 * 4);
    float* new_vars     = (float*)alloc((size_t)NV * 128 * 4);
    float* vdeg         = (float*)alloc((size_t)NR * 4);
    float* cdeg         = (float*)alloc((size_t)NC * 4);
    float* colsum       = (float*)alloc(128 * 4);
    float* sumsq        = (float*)alloc(4);
    float* pmean        = (float*)alloc(128 * 4);
    float* prs          = (float*)alloc(4);
    float* loss_sum     = (float*)alloc(4);
    unsigned short* CU  = (unsigned short*)alloc((size_t)NC * 512 * 2);
    unsigned short* HA  = (unsigned short*)alloc((size_t)NC * 256 * 2);
    unsigned short* HB  = (unsigned short*)alloc((size_t)NC * 256 * 2);
    unsigned short* INP = (unsigned short*)alloc((size_t)NA * 384 * 2);
    unsigned short* VB  = (unsigned short*)alloc((size_t)NV * 128 * 2);

    auto g1 = [](long long total) { return dim3((unsigned)((total + 255) / 256)); };
    auto gemm = [&](const unsigned short* A, const unsigned short* Wt, const float* bias,
                    void* O, int M, int K, int N, bool hidden) {
        dim3 grid(N / 128, M / 64), blk(256);
        if (hidden) k_gemm<true,  true ><<<grid, blk, 0, stream>>>(A, Wt, bias, O, M, K, N);
        else        k_gemm<false, false><<<grid, blk, 0, stream>>>(A, Wt, bias, O, M, K, N);
    };
    auto B = [&](int i) { return (const float*)d_in[wtab[i].idx + 1]; };

    // ---- weight conversion (once per launch; deterministic)
    for (int i = 0; i < 20; ++i) {
        long long tot = (long long)wtab[i].dout * wtab[i].dinP;
        k_wt<<<g1(tot), 256, 0, stream>>>((const float*)d_in[wtab[i].idx], WT[i],
                                          wtab[i].din, wtab[i].dout, wtab[i].dinP);
    }
    // ---- degrees
    hipMemsetAsync(vdeg, 0, (size_t)NR * 4, stream);
    hipMemsetAsync(cdeg, 0, (size_t)NC * 4, stream);
    k_count_deg<<<g1(NE), 256, 0, stream>>>(ROWS, COLS, vdeg, cdeg);
    k_rsqrt_deg<<<g1(NR), 256, 0, stream>>>(vdeg, NR);
    k_rsqrt_deg<<<g1(NC), 256, 0, stream>>>(cdeg, NC);
    // ---- init states, loss
    k_init_state<<<g1((long long)NV * 128), 256, 0, stream>>>(variables, (long long)NV * 128);
    k_init_state<<<g1((long long)NC * 128), 256, 0, stream>>>(clause_state, (long long)NC * 128);
    hipMemsetAsync(loss_sum, 0, 4, stream);

    for (int step = 0; step < ROUNDS; ++step) {
        // --- variables_query MLP: [NV,192] -> 384 -> 384 -> 128
        k_pack_qin<<<g1((long long)NV * 192), 256, 0, stream>>>(variables, INP, step);
        gemm(INP, WT[0], B(0), HA, NV, 192, 384, true);
        gemm(HA,  WT[1], B(1), HB, NV, 384, 384, true);
        gemm(HB,  WT[2], B(2), qbuf, NV, 384, 128, false);
        k_sigmoid<<<g1((long long)NV * 128), 256, 0, stream>>>(qbuf, (long long)NV * 128);
        k_pands<<<g1((long long)NA * 128), 256, 0, stream>>>(qbuf, I1, I2, p_ands);
        hipMemsetAsync(Sbuf, 0, (size_t)NC * 128 * 4, stream);
        k_scatter_S<<<g1((long long)NE * 128), 256, 0, stream>>>(ROWS, COLS, qbuf, p_ands, Sbuf);

        // --- var2ands MLP: [NA,256] -> 256 -> 256 -> 128
        k_pack_v2a<<<g1((long long)NA * 256), 256, 0, stream>>>(variables, I1, I2, INP);
        gemm(INP, WT[3], B(3), HA, NA, 256, 256, true);
        gemm(HA,  WT[4], B(4), HB, NA, 256, 256, true);
        gemm(HB,  WT[5], B(5), ands_data, NA, 256, 128, false);

        // --- var2clause MLP: [NV,128] -> 256 -> 256 -> 128
        k_cvt_bf<<<g1((long long)NV * 128), 256, 0, stream>>>(variables, VB, (long long)NV * 128);
        gemm(VB, WT[6], B(6), HA, NV, 128, 256, true);
        gemm(HA, WT[7], B(7), HB, NV, 256, 256, true);
        gemm(HB, WT[8], B(8), var_data, NV, 256, 128, false);

        // --- clause messages + clause_unit
        hipMemsetAsync(cmsg, 0, (size_t)NC * 128 * 4, stream);
        k_scatter_cmsg<<<g1((long long)NE * 128), 256, 0, stream>>>(ROWS, COLS, ZV, var_data, ands_data, cmsg);
        k_build_cu<<<g1((long long)NC * 128), 256, 0, stream>>>(clause_state, cmsg, cdeg, Sbuf, CU);

        // --- clause MLP: [NC,512] -> 256 -> 256 -> 256
        gemm(CU, WT[9],  B(9),  HA, NC, 512, 256, true);
        gemm(HA, WT[10], B(10), HB, NC, 256, 256, true);
        gemm(HB, WT[11], B(11), clause_data, NC, 256, 256, false);

        // --- clause_state = pairnorm(clause_data[:,128:])*0.25 + 0.1*clause_state
        hipMemsetAsync(colsum, 0, 128 * 4, stream);
        hipMemsetAsync(sumsq, 0, 4, stream);
        k_pn_reduce<<<NC, 128, 0, stream>>>(clause_data, 256, 128, colsum, sumsq);
        k_pn_stats<<<1, 128, 0, stream>>>(colsum, sumsq, pmean, prs, NC);
        k_pn_apply<<<g1((long long)NC * 128), 256, 0, stream>>>(clause_data, 256, 128, pmean, prs,
                                                                clause_state, (long long)NC * 128);

        // --- val = adj_mm(clause_data[:,:128]) * degree_w
        hipMemsetAsync(val, 0, (size_t)NR * 128 * 4, stream);
        k_scatter_val<<<g1((long long)NE * 128), 256, 0, stream>>>(ROWS, COLS, clause_data, val);
        k_scale_val<<<g1((long long)NR * 128), 256, 0, stream>>>(val, vdeg);

        // --- ands MLP: [NA,384] -> 256 -> 256 -> 128; scatter back into val
        k_pack_am<<<g1((long long)NA * 384), 256, 0, stream>>>(variables, val, I1, I2, INP);
        gemm(INP, WT[12], B(12), HA, NA, 384, 256, true);
        gemm(HA,  WT[13], B(13), HB, NA, 256, 256, true);
        gemm(HB,  WT[14], B(14), ands_value, NA, 256, 128, false);
        k_scatter_ands<<<g1((long long)NA * 128), 256, 0, stream>>>(ands_value, I1, I2, val);

        // --- update gate: [NV,256] -> 256 -> 256 -> 128
        k_pack_ug<<<g1((long long)NV * 256), 256, 0, stream>>>(variables, val, INP);
        gemm(INP, WT[15], B(15), HA, NV, 256, 256, true);
        gemm(HA,  WT[16], B(16), HB, NV, 256, 256, true);
        gemm(HB,  WT[17], B(17), new_vars, NV, 256, 128, false);
        hipMemsetAsync(colsum, 0, 128 * 4, stream);
        hipMemsetAsync(sumsq, 0, 4, stream);
        k_pn_reduce<<<NV, 128, 0, stream>>>(new_vars, 128, 0, colsum, sumsq);
        k_pn_stats<<<1, 128, 0, stream>>>(colsum, sumsq, pmean, prs, NV);
        k_pn_apply<<<g1((long long)NV * 128), 256, 0, stream>>>(new_vars, 128, 0, pmean, prs,
                                                                variables, (long long)NV * 128);

        // --- output MLP: [NV,128] -> 128 -> 128 -> 1 (+ xent loss)
        k_cvt_bf<<<g1((long long)NV * 128), 256, 0, stream>>>(variables, VB, (long long)NV * 128);
        gemm(VB, WT[18], B(18), HA, NV, 128, 128, true);
        gemm(HA, WT[19], B(19), HB, NV, 128, 128, true);
        k_out_final<<<NV / 256, 256, 0, stream>>>(HB, (const float*)d_in[40],
                                                  (const float*)d_in[41], LBL, out, loss_sum);
    }
    k_write_tail<<<1, 64, 0, stream>>>(loss_sum, out);
}
